// MyLSTM_52424370815292
// MI455X (gfx1250) — matrix operand order repaired and checked by reading, not compiled, not run on hardware
//
#include <hip/hip_runtime.h>
#include <cmath>

typedef __attribute__((ext_vector_type(16))) _Float16 v16h;
typedef __attribute__((ext_vector_type(8)))  float    v8f;

#define VOCAB 20000
#define EMB   300
#define EMBP  320      // padded to multiple of 32 (WMMA K granularity)
#define T_LEN 500
#define BATCH 128
#define UNITS 128
#define GCOLS 512      // 4*U gate columns

__device__ __forceinline__ float sigmoidf_(float x){ return 1.0f/(1.0f+__expf(-x)); }

// ---------------------------------------------------------------------------
// Pack a [K, 512] f32 weight into WMMA B-fragment-native f16 layout.
// Tile = (kt: 32 K-rows, nt: 16 N-cols). Within a tile, lane L owns
// 16 contiguous f16: element j -> W[kt*32 + (L>>4)*16 + j][nt*16 + (L&15)].
// Fragment load in the GEMMs is then one 32B contiguous load per lane.
// ---------------------------------------------------------------------------
__global__ void pack_w(const float* __restrict__ W, _Float16* __restrict__ out,
                       int K, int Kpad){
  int p = blockIdx.x*blockDim.x + threadIdx.x;
  int total = Kpad*GCOLS;
  if (p >= total) return;
  int tile = p >> 9;          // /512 elems per tile
  int r    = p & 511;
  int lane = r >> 4;
  int j    = r & 15;
  int kt   = tile >> 5;
  int nt   = tile & 31;
  int half = lane >> 4;
  int n    = nt*16 + (lane & 15);
  int k    = kt*32 + half*16 + j;
  float v = (k < K) ? W[k*GCOLS + n] : 0.0f;
  out[p] = (_Float16)v;
}

// ---------------------------------------------------------------------------
// Embedding gather -> time-major f16 X[T][B][EMBP], zero-padded K.
// ---------------------------------------------------------------------------
__global__ void embed_g(const int* __restrict__ tok, const float* __restrict__ emb,
                        _Float16* __restrict__ X){
  long p = (long)blockIdx.x*blockDim.x + threadIdx.x;
  long total = (long)T_LEN*BATCH*EMBP;
  if (p >= total) return;
  int e  = (int)(p % EMBP);
  long tb = p / EMBP;
  int b = (int)(tb % BATCH);
  int t = (int)(tb / BATCH);
  int id = tok[b*T_LEN + t];
  X[p] = (e < EMB) ? (_Float16)emb[(long)id*EMB + e] : (_Float16)0.0f;
}

// 16-bit A fragment: halves {0..15} relative to (row, kt*32 + half*16)
union AFrag { v16h v; uint4 q[3]; };   // q[0] = halves 0-7, q[1] = halves 8-15

// ---------------------------------------------------------------------------
// Batched input-projection GEMM: C[rows,512] f32 = A[rows,lda] f16 x Bpacked + bias.
// Block = 128 threads = 4 waves; block owns a 16-row strip; wave w owns
// cols [w*128, w*128+128) = 8 N-tiles. All 8 B fragments of a k-step are
// loaded into distinct registers first (one load clause), then 8 WMMAs issue
// back-to-back -> memory latency overlapped instead of load-wait-wmma chains.
// ---------------------------------------------------------------------------
__global__ __launch_bounds__(128)
void gemm_in(const _Float16* __restrict__ A, int lda, int ktiles,
             const _Float16* __restrict__ Bp,
             const float* __restrict__ bias,
             float* __restrict__ C){
  const int lane  = threadIdx.x & 31;
  const int wave  = threadIdx.x >> 5;
  const int mrow0 = blockIdx.x * 16;
  const int half  = lane >> 4;
  const int nlo   = lane & 15;

  const v8f vzero = {0.f,0.f,0.f,0.f,0.f,0.f,0.f,0.f};
  v8f acc[8];
  #pragma unroll
  for (int n=0;n<8;n++) acc[n] = vzero;

  for (int kt=0; kt<ktiles; ++kt){
    AFrag a;
    const uint4* ap = (const uint4*)(A + (long)(mrow0 + nlo)*lda + kt*32 + half*16);
    a.q[0] = ap[0];
    a.q[1] = ap[1];
    v16h bfr[8];
    #pragma unroll
    for (int n=0;n<8;n++){
      int nt = wave*8 + n;
      bfr[n] = *(const v16h*)(Bp + (long)(kt*32 + nt)*512 + lane*16);
    }
    #pragma unroll
    for (int n=0;n<8;n++)
      acc[n] = __builtin_amdgcn_wmma_f32_16x16x32_f16(
                 false, a.v, false, bfr[n], (short)0, acc[n], false, false);
  }
  #pragma unroll
  for (int n=0;n<8;n++){
    int col = (wave*8 + n)*16 + nlo;
    float bv = bias[col];
    #pragma unroll
    for (int e=0;e<8;e++){
      int row = mrow0 + e + half*8;              // WMMA C layout: VGPR e, half
      C[(long)row*GCOLS + col] = acc[n][e] + bv;
    }
  }
}

// ---------------------------------------------------------------------------
// Sequential recurrence. grid=2 (fwd/bwd direction per block), 512 threads
// (16 waves). The full packed Wh (128 KB) is staged in LDS once, alongside
// the 32 KB h state buffer (160 KB dynamic LDS, within the 320 KB WGP pool),
// so the 500 strictly-dependent steps run entirely out of LDS + registers:
// per step the only global traffic is the (prefetched) Z[t] slab.
// c state lives in registers, laid out to match the WMMA C fragment so the
// i,f,g,o pre-activations for one element are all in the SAME lane.
// Wave w: M-tile = w%8, u-tiles = (w/8)*4 .. +3, each with 4 gate tiles.
// ---------------------------------------------------------------------------
__global__ __launch_bounds__(512)
void lstm_seq(const float* __restrict__ Zf, const float* __restrict__ Zb,
              const _Float16* __restrict__ Whf, const _Float16* __restrict__ Whb,
              _Float16* __restrict__ Hseq,    // [T][128][256] or nullptr
              _Float16* __restrict__ Hlast){  // [128][256]    or nullptr
  extern __shared__ char smem_raw[];
  _Float16* hbuf  = (_Float16*)smem_raw;                          // 32 KB
  _Float16* whlds = (_Float16*)(smem_raw + BATCH*UNITS*2);        // 128 KB

  const int d = blockIdx.x;                   // 0 = forward, 1 = backward
  const float*    Z   = d ? Zb  : Zf;
  const _Float16* Whp = d ? Whb : Whf;
  const int dcol = d*UNITS;

  const int tid   = threadIdx.x;
  const int lane  = tid & 31;
  const int wave  = tid >> 5;
  const int mtile = wave & 7;
  const int ug0   = (wave >> 3) * 4;
  const int half  = lane >> 4;
  const int nlo   = lane & 15;

  // Stage packed Wh into LDS (b128 copies) and zero the h state.
  {
    const uint4* src = (const uint4*)Whp;
    uint4* dst = (uint4*)whlds;
    for (int i = tid; i < (UNITS*GCOLS)/8; i += 512) dst[i] = src[i];
  }
  for (int i = tid; i < BATCH*UNITS; i += 512) hbuf[i] = (_Float16)0.0f;

  float cst[4][8];
  #pragma unroll
  for (int g=0; g<4; g++)
    #pragma unroll
    for (int e=0; e<8; e++) cst[g][e] = 0.0f;
  __syncthreads();

  const v8f vzero = {0.f,0.f,0.f,0.f,0.f,0.f,0.f,0.f};

  for (int s = 0; s < T_LEN; ++s){
    const int t = d ? (T_LEN-1-s) : s;
    const float* zt = Z + (long)t*BATCH*GCOLS;

    // Prefetch next step's Z slab while this step computes.
    if (s+1 < T_LEN){
      const int tn = d ? (T_LEN-2-s) : (s+1);
      __builtin_prefetch(Z + (long)tn*BATCH*GCOLS + (long)tid*128, 0, 1);
    }

    v8f acc[4][4];
    #pragma unroll
    for (int g=0; g<4; g++)
      #pragma unroll
      for (int q=0; q<4; q++) acc[g][q] = vzero;

    #pragma unroll
    for (int kt=0; kt<4; ++kt){
      AFrag a;
      const uint4* ap = (const uint4*)(hbuf + (mtile*16 + nlo)*UNITS + kt*32 + half*16);
      a.q[0] = ap[0];
      a.q[1] = ap[1];
      #pragma unroll
      for (int g=0; g<4; g++){
        int utile = ug0 + g;
        v16h bfr[4];
        #pragma unroll
        for (int q=0; q<4; q++){
          int nt = q*8 + utile;   // col tile for gate q, unit-tile utile
          bfr[q] = *(const v16h*)(whlds + (long)(kt*32 + nt)*512 + lane*16);
        }
        #pragma unroll
        for (int q=0; q<4; q++)
          acc[g][q] = __builtin_amdgcn_wmma_f32_16x16x32_f16(
                        false, a.v, false, bfr[q], (short)0, acc[g][q], false, false);
      }
    }
    __syncthreads();   // all hbuf reads complete before anyone overwrites h

    #pragma unroll
    for (int g=0; g<4; g++){
      const int u = (ug0+g)*16 + nlo;
      #pragma unroll
      for (int e=0; e<8; e++){
        const int row = mtile*16 + e + half*8;
        const float* zr = zt + (long)row*GCOLS;
        float zi = acc[g][0][e] + zr[      u];
        float zfg= acc[g][1][e] + zr[128 + u];
        float zg = acc[g][2][e] + zr[256 + u];
        float zo = acc[g][3][e] + zr[384 + u];
        float c  = sigmoidf_(zfg)*cst[g][e] + sigmoidf_(zi)*tanhf(zg);
        cst[g][e] = c;
        float h  = sigmoidf_(zo)*tanhf(c);
        _Float16 hh = (_Float16)h;
        hbuf[row*UNITS + u] = hh;
        if (Hseq)
          Hseq[(long)t*BATCH*(2*UNITS) + (long)row*(2*UNITS) + dcol + u] = hh;
      }
    }
    __syncthreads();   // new h visible before next step's fragment loads
  }

  if (Hlast){
    for (int i = tid; i < BATCH*UNITS; i += 512){
      int row = i >> 7, u = i & 127;
      Hlast[row*(2*UNITS) + dcol + u] = hbuf[i];
    }
  }
}

// ---------------------------------------------------------------------------
// Dense head: relu(h @ Wd1 + bd1) @ Wd2 + bd2 -> sigmoid. Tiny: scalar VALU.
// ---------------------------------------------------------------------------
__global__ __launch_bounds__(32)
void head_k(const _Float16* __restrict__ H, const float* __restrict__ Wd1,
            const float* __restrict__ bd1, const float* __restrict__ Wd2,
            const float* __restrict__ bd2, float* __restrict__ out){
  __shared__ float hid[32];
  const int b = blockIdx.x;
  const int j = threadIdx.x;
  float s = bd1[j];
  for (int k=0;k<256;k++) s += (float)H[b*256+k]*Wd1[k*32+j];
  hid[j] = fmaxf(s, 0.0f);
  __syncthreads();
  if (j < 6){
    float l = bd2[j];
    for (int k=0;k<32;k++) l += hid[k]*Wd2[k*6+j];
    out[b*6+j] = 1.0f/(1.0f+__expf(-l));
  }
}

// ---------------------------------------------------------------------------
extern "C" void kernel_launch(void* const* d_in, const int* in_sizes, int n_in,
                              void* d_out, int out_size, void* d_ws, size_t ws_size,
                              hipStream_t stream){
  (void)in_sizes; (void)n_in; (void)out_size; (void)ws_size;
  const int*   tokens = (const int*)  d_in[0];
  const float* emb    = (const float*)d_in[1];
  const float* W1f_i  = (const float*)d_in[2];
  const float* W1f_h  = (const float*)d_in[3];
  const float* b1f    = (const float*)d_in[4];
  const float* W1b_i  = (const float*)d_in[5];
  const float* W1b_h  = (const float*)d_in[6];
  const float* b1b    = (const float*)d_in[7];
  const float* W2f_i  = (const float*)d_in[8];
  const float* W2f_h  = (const float*)d_in[9];
  const float* b2f    = (const float*)d_in[10];
  const float* W2b_i  = (const float*)d_in[11];
  const float* W2b_h  = (const float*)d_in[12];
  const float* b2b    = (const float*)d_in[13];
  const float* Wd1    = (const float*)d_in[14];
  const float* bd1    = (const float*)d_in[15];
  const float* Wd2    = (const float*)d_in[16];
  const float* bd2    = (const float*)d_in[17];
  float* out = (float*)d_out;

  char* ws = (char*)d_ws;
  size_t off = 0;
  auto carve = [&](size_t bytes)->char*{
    char* p = ws + off; off += (bytes + 255) & ~(size_t)255; return p;
  };
  _Float16* X    = (_Float16*)carve((size_t)T_LEN*BATCH*EMBP*2);   // 41 MB
  _Float16* p1fi = (_Float16*)carve((size_t)EMBP*GCOLS*2);
  _Float16* p1bi = (_Float16*)carve((size_t)EMBP*GCOLS*2);
  _Float16* p2fi = (_Float16*)carve((size_t)256 *GCOLS*2);
  _Float16* p2bi = (_Float16*)carve((size_t)256 *GCOLS*2);
  _Float16* ph1f = (_Float16*)carve((size_t)UNITS*GCOLS*2);
  _Float16* ph1b = (_Float16*)carve((size_t)UNITS*GCOLS*2);
  _Float16* ph2f = (_Float16*)carve((size_t)UNITS*GCOLS*2);
  _Float16* ph2b = (_Float16*)carve((size_t)UNITS*GCOLS*2);
  float*    ZA   = (float*)   carve((size_t)T_LEN*BATCH*GCOLS*4);  // 131 MB, reused L1->L2
  float*    ZB   = (float*)   carve((size_t)T_LEN*BATCH*GCOLS*4);  // 131 MB, reused L1->L2
  _Float16* H1   = (_Float16*)carve((size_t)T_LEN*BATCH*256*2);    // 33 MB
  _Float16* HL   = (_Float16*)carve((size_t)BATCH*256*2);

  auto pack = [&](const float* W, _Float16* o, int K, int Kpad){
    int total = Kpad*GCOLS;
    pack_w<<<(total+255)/256, 256, 0, stream>>>(W, o, K, Kpad);
  };
  pack(W1f_i, p1fi, EMB, EMBP);
  pack(W1b_i, p1bi, EMB, EMBP);
  pack(W2f_i, p2fi, 256, 256);
  pack(W2b_i, p2bi, 256, 256);
  pack(W1f_h, ph1f, UNITS, UNITS);
  pack(W1b_h, ph1b, UNITS, UNITS);
  pack(W2f_h, ph2f, UNITS, UNITS);
  pack(W2b_h, ph2b, UNITS, UNITS);

  { long tot = (long)T_LEN*BATCH*EMBP;
    embed_g<<<(int)((tot+255)/256), 256, 0, stream>>>(tokens, emb, X); }

  const int rowTiles = (T_LEN*BATCH)/16;   // 4000
  const int lstmLds  = (BATCH*UNITS + UNITS*GCOLS)*2;   // 160 KB dynamic LDS

  // Layer 1 input projections (bias folded in)
  gemm_in<<<rowTiles, 128, 0, stream>>>(X, EMBP, EMBP/32, p1fi, b1f, ZA);
  gemm_in<<<rowTiles, 128, 0, stream>>>(X, EMBP, EMBP/32, p1bi, b1b, ZB);
  // Layer 1 recurrence (both directions concurrently)
  lstm_seq<<<2, 512, lstmLds, stream>>>(ZA, ZB, ph1f, ph1b, H1, nullptr);
  // Layer 2 input projections (reuse Z buffers — layer-1 Z is dead now)
  gemm_in<<<rowTiles, 128, 0, stream>>>(H1, 256, 8, p2fi, b2f, ZA);
  gemm_in<<<rowTiles, 128, 0, stream>>>(H1, 256, 8, p2bi, b2b, ZB);
  // Layer 2 recurrence, final states only
  lstm_seq<<<2, 512, lstmLds, stream>>>(ZA, ZB, ph2f, ph2b, nullptr, HL);
  // Head
  head_k<<<BATCH, 32, 0, stream>>>(HL, Wd1, bd1, Wd2, bd2, out);
}
